// GNN_78013785964837
// MI455X (gfx1250) — matrix operand order repaired and checked by reading, not compiled, not run on hardware
//
#include <hip/hip_runtime.h>

#define BB 8
#define LL 2048
#define EE 256
#define HH 4
#define DD 64

typedef __bf16 bf16_t;
typedef __attribute__((ext_vector_type(16))) __bf16 v16bf;
typedef __attribute__((ext_vector_type(8)))  __bf16 v8bf;
typedef __attribute__((ext_vector_type(8)))  float  v8f;

// Build a 16x32 bf16 A-operand vector from two 16-byte chunks.
__device__ __forceinline__ v16bf load_pair16(const bf16_t* lo, const bf16_t* hi) {
  v8bf l = *(const v8bf*)lo;
  v8bf h = *(const v8bf*)hi;
  v16bf r;
#pragma unroll
  for (int i = 0; i < 8; ++i) { r[i] = l[i]; r[i + 8] = h[i]; }
  return r;
}

__device__ __forceinline__ v8f wmma_bf16(v16bf a, v16bf b, v8f c) {
  return __builtin_amdgcn_wmma_f32_16x16x32_bf16(false, a, false, b, (short)0, c,
                                                 false, false);
}

// ------------- x: fp32 -> bf16, producing BOTH row-major and transposed ------
// xb [b][l][e], xbT [b][e][l]. 64x64 tiles through LDS, fully coalesced.
__global__ __launch_bounds__(256) void k_cvt_x(const float* __restrict__ x,
                                               bf16_t* __restrict__ xb,
                                               bf16_t* __restrict__ xbT) {
  __shared__ bf16_t tile[64][65];  // [e_local][l_local], padded
  int bid = blockIdx.x;
  int b  = bid >> 7;        // (L/64)*(E/64) = 32*4 = 128 tiles per batch
  int lt = (bid >> 2) & 31; // l tile
  int et = bid & 3;         // e tile
  int c  = threadIdx.x & 63;
  int r0 = threadIdx.x >> 6;
#pragma unroll
  for (int j = 0; j < 16; ++j) {
    int r = r0 + j * 4;  // l_local
    size_t idx = ((size_t)(b * LL) + lt * 64 + r) * EE + et * 64 + c;
    bf16_t bv = (bf16_t)x[idx];
    xb[idx] = bv;
    tile[c][r] = bv;
  }
  __syncthreads();
#pragma unroll
  for (int j = 0; j < 16; ++j) {
    int r = r0 + j * 4;  // e_local
    xbT[((size_t)(b * EE) + et * 64 + r) * LL + lt * 64 + c] = tile[r][c];
  }
}

__global__ void k_cvt(const float* __restrict__ src, bf16_t* __restrict__ dst, int n) {
  int i = blockIdx.x * blockDim.x + threadIdx.x;
  if (i < n) dst[i] = (bf16_t)src[i];
}

// dst[n][k] = src[k][n]  (transpose W_gnn so B-operand reads are contiguous)
__global__ void k_cvt_T(const float* __restrict__ src, bf16_t* __restrict__ dst) {
  int i = blockIdx.x * blockDim.x + threadIdx.x;
  int k = i / EE, n = i % EE;
  dst[n * EE + k] = (bf16_t)src[k * EE + n];
}

// ---------------- QK projection: qk = x @ Wqk^T + b ; q scaled by 1/sqrt(D) ---
__global__ __launch_bounds__(128) void k_qk(const bf16_t* __restrict__ xb,
                                            const bf16_t* __restrict__ wqk,
                                            const float* __restrict__ bias,
                                            bf16_t* __restrict__ qb,
                                            bf16_t* __restrict__ kbuf) {
  int wave = blockIdx.x * 4 + (threadIdx.x >> 5);
  int lane = threadIdx.x & 31;
  int lr = lane & 15, g = lane >> 4;
  int tn = wave & 31;   // N = 512 -> 32 n-tiles
  int tm = wave >> 5;   // M = 16384 -> 1024 m-tiles
  const bf16_t* arow = xb + (size_t)(tm * 16 + lr) * EE;
  const bf16_t* brow = wqk + (size_t)(tn * 16 + lr) * EE;
  v8f c = {};
#pragma unroll
  for (int k0 = 0; k0 < EE; k0 += 32) {
    v16bf a = load_pair16(arow + k0 + g * 16, arow + k0 + g * 16 + 8);
    v16bf b = *(const v16bf*)(brow + k0 + g * 16);
    c = wmma_bf16(a, b, c);
  }
  int n = tn * 16 + lr;
  float bv = bias[n];
  bool isq = (n < EE);
  float scale = isq ? 0.125f : 1.0f;  // 1/sqrt(64)
  bf16_t* dst = isq ? (qb + n) : (kbuf + (n - EE));
#pragma unroll
  for (int v = 0; v < 8; ++v) {
    int m = tm * 16 + v + g * 8;
    dst[(size_t)m * EE] = (bf16_t)((c[v] + bv) * scale);
  }
}

// ------- fused flash attention + head-average + (P @ x), x read transposed ----
// 64-key chunks: softmax bookkeeping amortized over 40 WMMAs per chunk.
__global__ __launch_bounds__(128) void k_attn(const bf16_t* __restrict__ qb,
                                              const bf16_t* __restrict__ kbuf,
                                              const bf16_t* __restrict__ xbT,
                                              bf16_t* __restrict__ aggb) {
  __shared__ float  acc[16 * EE];       // head-averaged output tile, 16 KB
  __shared__ bf16_t pbuf[4 * 16 * 64];  // per-wave P bounce (16x64), 8 KB

  int tid = threadIdx.x;
  int lane = tid & 31;
  int lr = lane & 15, g = lane >> 4;
  int h = tid >> 5;             // wave -> head
  int batch = blockIdx.x >> 7;  // 128 query tiles per batch
  int qt = blockIdx.x & 127;

  for (int i = tid; i < 16 * EE; i += 128) acc[i] = 0.f;
  __syncthreads();

  // Q tile A-operands for this head, kept in registers for the whole loop
  const bf16_t* qrow = qb + ((size_t)(batch * LL + qt * 16 + lr) * EE + h * DD);
  v16bf aq0 = load_pair16(qrow + g * 16, qrow + g * 16 + 8);       // d = 0..31
  v16bf aq1 = load_pair16(qrow + 32 + g * 16, qrow + 40 + g * 16);  // d = 32..63

  float mrun[8], lrun[8];
#pragma unroll
  for (int v = 0; v < 8; ++v) { mrun[v] = -1e30f; lrun[v] = 0.f; }
  v8f O[16] = {};

  bf16_t* pb = pbuf + h * (16 * 64);
  const bf16_t* xtb = xbT + (size_t)(batch * EE) * LL;  // [e][key] slab

  for (int kc = 0; kc < LL / 64; ++kc) {
    // scores S = Qs * K^T for 64 keys (four 16x16 tiles)
    size_t keybase = (size_t)batch * LL + kc * 64;
    const bf16_t* krow0 = kbuf + (keybase + lr) * EE + h * DD + g * 16;
    const bf16_t* krow1 = krow0 + 16 * EE;
    const bf16_t* krow2 = krow0 + 32 * EE;
    const bf16_t* krow3 = krow0 + 48 * EE;
    v8f s0 = {}, s1 = {}, s2 = {}, s3 = {};
    s0 = wmma_bf16(aq0, *(const v16bf*)(krow0), s0);
    s0 = wmma_bf16(aq1, *(const v16bf*)(krow0 + 32), s0);
    s1 = wmma_bf16(aq0, *(const v16bf*)(krow1), s1);
    s1 = wmma_bf16(aq1, *(const v16bf*)(krow1 + 32), s1);
    s2 = wmma_bf16(aq0, *(const v16bf*)(krow2), s2);
    s2 = wmma_bf16(aq1, *(const v16bf*)(krow2 + 32), s2);
    s3 = wmma_bf16(aq0, *(const v16bf*)(krow3), s3);
    s3 = wmma_bf16(aq1, *(const v16bf*)(krow3 + 32), s3);

    if (kc + 1 < LL / 64) {  // hint next chunk into cache (global_prefetch_b8)
      __builtin_prefetch((const void*)(krow0 + 64 * EE), 0, 1);
      __builtin_prefetch((const void*)(xtb + (size_t)(h * DD + lr) * LL + kc * 64 + 64), 0, 1);
    }

    // online softmax (row stats live in 16-lane halves matching C layout)
    float alpha[8];
#pragma unroll
    for (int v = 0; v < 8; ++v) {
      float cm = fmaxf(fmaxf(s0[v], s1[v]), fmaxf(s2[v], s3[v]));
      cm = fmaxf(cm, __shfl_xor(cm, 1));
      cm = fmaxf(cm, __shfl_xor(cm, 2));
      cm = fmaxf(cm, __shfl_xor(cm, 4));
      cm = fmaxf(cm, __shfl_xor(cm, 8));
      float mnew = fmaxf(mrun[v], cm);
      alpha[v] = __expf(mrun[v] - mnew);
      mrun[v] = mnew;
      float p0 = __expf(s0[v] - mnew);
      float p1 = __expf(s1[v] - mnew);
      float p2 = __expf(s2[v] - mnew);
      float p3 = __expf(s3[v] - mnew);
      s0[v] = p0; s1[v] = p1; s2[v] = p2; s3[v] = p3;
      float rs = (p0 + p1) + (p2 + p3);
      rs += __shfl_xor(rs, 1);
      rs += __shfl_xor(rs, 2);
      rs += __shfl_xor(rs, 4);
      rs += __shfl_xor(rs, 8);
      lrun[v] = lrun[v] * alpha[v] + rs;
    }
#pragma unroll
    for (int nt = 0; nt < 16; ++nt)
#pragma unroll
      for (int v = 0; v < 8; ++v) O[nt][v] *= alpha[v];

    // C-layout -> A-layout for P (16x64 tile) via per-wave LDS bounce
#pragma unroll
    for (int v = 0; v < 8; ++v) {
      int row = v + g * 8;
      pb[row * 64 + lr]      = (bf16_t)s0[v];
      pb[row * 64 + 16 + lr] = (bf16_t)s1[v];
      pb[row * 64 + 32 + lr] = (bf16_t)s2[v];
      pb[row * 64 + 48 + lr] = (bf16_t)s3[v];
    }
    asm volatile("" ::: "memory");
    v16bf pa0 = load_pair16(pb + lr * 64 + g * 16,      pb + lr * 64 + 8 + g * 16);
    v16bf pa1 = load_pair16(pb + lr * 64 + 32 + g * 16, pb + lr * 64 + 40 + g * 16);

    // O += P @ x_chunk ; B operand = contiguous 32B per lane from xbT
#pragma unroll
    for (int nt = 0; nt < 16; ++nt) {
      const bf16_t* xrow = xtb + (size_t)(nt * 16 + lr) * LL + kc * 64;
      v16bf xv0 = *(const v16bf*)(xrow + g * 16);
      v16bf xv1 = *(const v16bf*)(xrow + 32 + g * 16);
      O[nt] = wmma_bf16(pa0, xv0, O[nt]);
      O[nt] = wmma_bf16(pa1, xv1, O[nt]);
    }
  }

  // normalize, fold in 1/H, head-sum into shared accumulator
  float inv[8];
#pragma unroll
  for (int v = 0; v < 8; ++v) inv[v] = 0.25f / lrun[v];
#pragma unroll
  for (int nt = 0; nt < 16; ++nt)
#pragma unroll
    for (int v = 0; v < 8; ++v) {
      int r = v + g * 8;
      atomicAdd(&acc[r * EE + nt * 16 + lr], O[nt][v] * inv[v]);
    }
  __syncthreads();
  size_t obase = (size_t)(batch * LL + qt * 16) * EE;
  for (int i = tid; i < 16 * EE; i += 128) aggb[obase + i] = (bf16_t)acc[i];
}

// ---------------- final GEMM: out = agg @ W_gnn + B_gnn (fp32 out) -----------
__global__ __launch_bounds__(128) void k_out(const bf16_t* __restrict__ aggb,
                                             const bf16_t* __restrict__ wgT,
                                             const float* __restrict__ bg,
                                             float* __restrict__ out) {
  int wave = blockIdx.x * 4 + (threadIdx.x >> 5);
  int lane = threadIdx.x & 31;
  int lr = lane & 15, g = lane >> 4;
  int tn = wave & 15;   // N = 256 -> 16 n-tiles
  int tm = wave >> 4;   // M = 16384 -> 1024 m-tiles
  const bf16_t* arow = aggb + (size_t)(tm * 16 + lr) * EE;
  const bf16_t* brow = wgT + (size_t)(tn * 16 + lr) * EE;
  v8f c = {};
#pragma unroll
  for (int k0 = 0; k0 < EE; k0 += 32) {
    v16bf a = load_pair16(arow + k0 + g * 16, arow + k0 + g * 16 + 8);
    v16bf b = *(const v16bf*)(brow + k0 + g * 16);
    c = wmma_bf16(a, b, c);
  }
  int n = tn * 16 + lr;
  float bv = bg[n];
#pragma unroll
  for (int v = 0; v < 8; ++v) {
    int m = tm * 16 + v + g * 8;
    out[(size_t)m * EE + n] = c[v] + bv;
  }
}

extern "C" void kernel_launch(void* const* d_in, const int* in_sizes, int n_in,
                              void* d_out, int out_size, void* d_ws, size_t ws_size,
                              hipStream_t stream) {
  const float* x  = (const float*)d_in[0];   // [B,L,E]
  const float* wi = (const float*)d_in[1];   // [3E,E]
  const float* bi = (const float*)d_in[2];   // [3E]
  const float* wg = (const float*)d_in[3];   // [E,E]
  const float* bg = (const float*)d_in[4];   // [E]
  float* out = (float*)d_out;

  char* ws = (char*)d_ws;
  size_t off = 0;
  bf16_t* xb  = (bf16_t*)(ws + off); off += (size_t)BB * LL * EE * 2;
  bf16_t* xbT = (bf16_t*)(ws + off); off += (size_t)BB * LL * EE * 2;
  bf16_t* qb  = (bf16_t*)(ws + off); off += (size_t)BB * LL * EE * 2;
  bf16_t* kb  = (bf16_t*)(ws + off); off += (size_t)BB * LL * EE * 2;
  bf16_t* agg = (bf16_t*)(ws + off); off += (size_t)BB * LL * EE * 2;
  bf16_t* wqk = (bf16_t*)(ws + off); off += (size_t)2 * EE * EE * 2;
  bf16_t* wgT = (bf16_t*)(ws + off); off += (size_t)EE * EE * 2;
  (void)off; (void)ws_size; (void)in_sizes; (void)n_in; (void)out_size;

  k_cvt_x<<<BB * (LL / 64) * (EE / 64), 256, 0, stream>>>(x, xb, xbT);  // 1024 blocks
  k_cvt<<<(2 * EE * EE) / 256, 256, 0, stream>>>(wi, wqk, 2 * EE * EE);
  k_cvt_T<<<(EE * EE) / 256, 256, 0, stream>>>(wg, wgT);
  k_qk<<<8192, 128, 0, stream>>>(xb, wqk, bi, qb, kb);          // 32768 tiles / 4 waves
  k_attn<<<BB * (LL / 16), 128, 0, stream>>>(qb, kb, xbT, agg); // 1024 WGs
  k_out<<<4096, 128, 0, stream>>>(agg, wgT, bg, out);           // 16384 tiles / 4 waves
}
